// KernelNN_2954937499677
// MI455X (gfx1250) — compile-verified, matrix operand order checked by reading, not run on hardware
//
#include <hip/hip_runtime.h>
#include <hip/hip_bf16.h>

// ---------------------------------------------------------------------------
// Types for CDNA5 WMMA (wave32)
// ---------------------------------------------------------------------------
typedef __attribute__((ext_vector_type(16))) __bf16        v16bf;
typedef __attribute__((ext_vector_type(8)))  float         v8f;
typedef __attribute__((ext_vector_type(4))) unsigned int   uint4v;
typedef __attribute__((ext_vector_type(8))) unsigned int   uint8v;

#define W_STEPS   8
#define F_DIM     32
#define KW_DIM    256
#define KIN_DIM   6
#define TILE_E    64          // edges per block (4 WMMA M-tiles, B-frag reuse x4)
#define M_TILES   4
#define LDS_STRIDE 264        // 256 + 8 ushort pad -> 528B row stride (16B aligned, bank-spread)

// ---------------------------------------------------------------------------
// bf16 <-> f32 helpers (round-to-nearest-even)
// ---------------------------------------------------------------------------
__device__ __forceinline__ unsigned short f2bf_u(float f) {
    unsigned int u = __builtin_bit_cast(unsigned int, f);
    u += 0x7FFFu + ((u >> 16) & 1u);
    return (unsigned short)(u >> 16);
}
__device__ __forceinline__ float bf2f(unsigned short s) {
    unsigned int u = ((unsigned int)s) << 16;
    return __builtin_bit_cast(float, u);
}
__device__ __forceinline__ float sigmoidf_(float x) {
    return 1.0f / (1.0f + expf(-x));
}

// Build a v16bf A/B fragment for v_wmma_f32_16x16x32_bf16.
// 16-bit operand layout (ISA 7.12.2): per lane, elements 0..7 = K kb..kb+7,
// elements 8..15 = K kb+16..kb+23 (lanes 16..31 use kb += 8, handled by caller).
// `base` points at element kb of a contiguous-in-K row; must be 16B aligned.
__device__ __forceinline__ v16bf load_frag(const unsigned short* base) {
    uint4v lo = *(const uint4v*)(base);        // K kb .. kb+7
    uint4v hi = *(const uint4v*)(base + 16);   // K kb+16 .. kb+23
    uint8v c;
    c[0] = lo[0]; c[1] = lo[1]; c[2] = lo[2]; c[3] = lo[3];
    c[4] = hi[0]; c[5] = hi[1]; c[6] = hi[2]; c[7] = hi[3];
    return __builtin_bit_cast(v16bf, c);
}

// ---------------------------------------------------------------------------
// Kernel: f32 -> bf16 weight conversion (kW2, kW3 staged once, live in L2)
// ---------------------------------------------------------------------------
__global__ void cvt_bf16_kernel(const float* __restrict__ in,
                                unsigned short* __restrict__ out, int n) {
    int i = blockIdx.x * blockDim.x + threadIdx.x;
    if (i < n) out[i] = f2bf_u(in[i]);
}

// ---------------------------------------------------------------------------
// Kernel: LSTM(3,3) over 8 window steps + fc(3->32) + ReLU. Thread per node.
// ---------------------------------------------------------------------------
__global__ void lstm_fc_kernel(const float* __restrict__ xpos,
                               const float* __restrict__ W_ih,
                               const float* __restrict__ W_hh,
                               const float* __restrict__ b_ih,
                               const float* __restrict__ b_hh,
                               const float* __restrict__ W_fc,
                               const float* __restrict__ b_fc,
                               float* __restrict__ x_out, int N) {
    int n = blockIdx.x * blockDim.x + threadIdx.x;
    if (n >= N) return;
    float h[3] = {0.f, 0.f, 0.f};
    float c[3] = {0.f, 0.f, 0.f};
    for (int t = 0; t < W_STEPS; ++t) {
        float xt[3];
        #pragma unroll
        for (int d = 0; d < 3; ++d) xt[d] = xpos[((size_t)t * N + n) * 3 + d];
        float g[12];
        #pragma unroll
        for (int r = 0; r < 12; ++r) {
            float s = b_ih[r] + b_hh[r];
            #pragma unroll
            for (int k = 0; k < 3; ++k)
                s += xt[k] * W_ih[r * 3 + k] + h[k] * W_hh[r * 3 + k];
            g[r] = s;
        }
        #pragma unroll
        for (int j = 0; j < 3; ++j) {      // torch gate order: i, f, g, o
            float ig = sigmoidf_(g[j]);
            float fg = sigmoidf_(g[3 + j]);
            float gg = tanhf(g[6 + j]);
            float og = sigmoidf_(g[9 + j]);
            c[j] = fg * c[j] + ig * gg;
            h[j] = og * tanhf(c[j]);
        }
    }
    #pragma unroll
    for (int o = 0; o < F_DIM; ++o) {
        float s = b_fc[o];
        #pragma unroll
        for (int k = 0; k < 3; ++k) s += h[k] * W_fc[o * 3 + k];
        x_out[(size_t)n * F_DIM + o] = fmaxf(s, 0.f);
    }
}

// ---------------------------------------------------------------------------
// Kernel: in-degree counts for scatter-mean
// ---------------------------------------------------------------------------
__global__ void degree_kernel(const int* __restrict__ eidx, float* __restrict__ cnt, int E) {
    int i = blockIdx.x * blockDim.x + threadIdx.x;
    if (i < E) atomicAdd(&cnt[eidx[E + i]], 1.0f);
}

// ---------------------------------------------------------------------------
// Kernel: edge-conditioned kernel MLP -> w (bf16 [E,1024]) via WMMA bf16.
// Block = 256 threads (8 waves), 64 edges per block (4 M-tiles of 16).
// B fragments are held in registers and reused across the 4 M-tiles, cutting
// bf16 weight (kW2/kW3) L2 traffic 4x vs. a 16-edge tile.
//   L1 (K=6, VALU)         -> hk1 bf16 LDS [64,256]
//   L2 (WMMA, 64/wave)     -> hk2 bf16 LDS [64,256]
//   L3 (WMMA, 256/wave)    -> w bf16 global
// ---------------------------------------------------------------------------
__global__ __launch_bounds__(256) void edge_mlp_kernel(
    const float* __restrict__ edge_attr,
    const float* __restrict__ kW1, const float* __restrict__ kb1,
    const float* __restrict__ kb2, const float* __restrict__ kb3,
    const unsigned short* __restrict__ kW2b,
    const unsigned short* __restrict__ kW3b,
    unsigned short* __restrict__ wout, int E) {

    __shared__ __align__(16) float          ea_s[TILE_E][8];
    __shared__ __align__(16) unsigned short hk1_s[TILE_E][LDS_STRIDE];
    __shared__ __align__(16) unsigned short hk2_s[TILE_E][LDS_STRIDE];

    const int tid = threadIdx.x;
    const int e0  = blockIdx.x * TILE_E;

    // ---- stage edge attributes -------------------------------------------
    for (int i = tid; i < TILE_E * KIN_DIM; i += 256) {
        int e = i / KIN_DIM, k = i % KIN_DIM;
        ea_s[e][k] = (e0 + e < E) ? edge_attr[(size_t)(e0 + e) * KIN_DIM + k] : 0.f;
    }
    __syncthreads();

    // ---- layer 1: [64,6] @ [6,256] + ReLU -> bf16 LDS --------------------
    {
        const int el = tid >> 2;        // 4 threads per edge
        const int n0 = tid & 3;
        #pragma unroll
        for (int j = 0; j < 64; ++j) {
            int n = n0 + j * 4;
            float s = kb1[n];
            #pragma unroll
            for (int k = 0; k < KIN_DIM; ++k) s += ea_s[el][k] * kW1[n * KIN_DIM + k];
            hk1_s[el][n] = f2bf_u(fmaxf(s, 0.f));
        }
    }
    __syncthreads();

    const int wv    = tid >> 5;
    const int lane  = tid & 31;
    const int rowA  = lane & 15;            // A: edge row / B,D: tile col
    const int koff  = (lane & 16) ? 8 : 0;  // K half select for hi lanes
    const int mbase = (lane & 16) ? 8 : 0;  // D: M base for hi lanes

    // ---- layer 2: [64,256] @ [256,256] + ReLU -> bf16 LDS ---------------
    for (int tt = 0; tt < 2; ++tt) {
        const int tn = wv + tt * 8;             // 16 column tiles / 8 waves
        v8f acc[M_TILES] = {};
        #pragma unroll
        for (int k0 = 0; k0 < 8; ++k0) {
            const int kb = k0 * 32 + koff;
            v16bf b = load_frag(&kW2b[(size_t)(tn * 16 + rowA) * KW_DIM + kb]);
            #pragma unroll
            for (int m = 0; m < M_TILES; ++m) {
                v16bf a = load_frag(&hk1_s[m * 16 + rowA][kb]);
                acc[m] = __builtin_amdgcn_wmma_f32_16x16x32_bf16(
                    false, a, false, b, (short)0, acc[m], false, false);
            }
        }
        const int col  = tn * 16 + rowA;
        const float bs = kb2[col];
        #pragma unroll
        for (int m = 0; m < M_TILES; ++m) {
            #pragma unroll
            for (int j = 0; j < 8; ++j) {
                hk2_s[m * 16 + mbase + j][col] = f2bf_u(fmaxf(acc[m][j] + bs, 0.f));
            }
        }
    }
    __syncthreads();

    // ---- layer 3: [64,256] @ [256,1024] -> w bf16 global -----------------
    for (int tt = 0; tt < 8; ++tt) {
        const int tn = wv * 8 + tt;             // 64 column tiles / 8 waves
        v8f acc[M_TILES] = {};
        #pragma unroll
        for (int k0 = 0; k0 < 8; ++k0) {
            const int kb = k0 * 32 + koff;
            v16bf b = load_frag(&kW3b[(size_t)(tn * 16 + rowA) * KW_DIM + kb]);
            #pragma unroll
            for (int m = 0; m < M_TILES; ++m) {
                v16bf a = load_frag(&hk2_s[m * 16 + rowA][kb]);
                acc[m] = __builtin_amdgcn_wmma_f32_16x16x32_bf16(
                    false, a, false, b, (short)0, acc[m], false, false);
            }
        }
        const int o    = tn * 16 + rowA;        // flat output column in [0,1024)
        const float bs = kb3[o];
        #pragma unroll
        for (int m = 0; m < M_TILES; ++m) {
            #pragma unroll
            for (int j = 0; j < 8; ++j) {
                int e = e0 + m * 16 + mbase + j;
                if (e < E) wout[(size_t)e * 1024 + o] = f2bf_u(acc[m][j] + bs);
            }
        }
    }
}

// ---------------------------------------------------------------------------
// Kernel: per-edge message msg = x[src] @ w[e], scatter-add to agg[dst].
// One wave per edge; shfl-broadcast x[src], coalesced bf16 w reads.
// ---------------------------------------------------------------------------
__global__ __launch_bounds__(256) void conv_msg_kernel(
    const float* __restrict__ x, const unsigned short* __restrict__ wbf,
    const int* __restrict__ eidx, float* __restrict__ agg, int E) {
    int gid  = blockIdx.x * blockDim.x + threadIdx.x;
    int e    = gid >> 5;
    int lane = gid & 31;
    if (e >= E) return;
    const unsigned short* wp = wbf + (size_t)e * 1024;
    __builtin_prefetch(wp + lane * 32, 0, 0);   // -> global_prefetch_b8: start 2KB tile fetch early
    int src = eidx[e];
    int dst = eidx[E + e];
    float xv  = x[(size_t)src * F_DIM + lane];
    float acc = 0.f;
    #pragma unroll 8
    for (int i = 0; i < F_DIM; ++i) {
        float xi = __shfl(xv, i, 32);
        acc += xi * bf2f(wp[i * F_DIM + lane]);
    }
    atomicAdd(&agg[(size_t)dst * F_DIM + lane], acc);
}

// ---------------------------------------------------------------------------
// Kernel: x_new = ReLU(agg/cnt + x@root + bias). One wave per node.
// ---------------------------------------------------------------------------
__global__ __launch_bounds__(256) void combine_kernel(
    const float* __restrict__ x, const float* __restrict__ agg,
    const float* __restrict__ cnt, const float* __restrict__ root,
    const float* __restrict__ bias, float* __restrict__ xn, int N) {
    int gid  = blockIdx.x * blockDim.x + threadIdx.x;
    int n    = gid >> 5;
    int lane = gid & 31;
    if (n >= N) return;
    float xv = x[(size_t)n * F_DIM + lane];
    float s  = agg[(size_t)n * F_DIM + lane] / fmaxf(cnt[n], 1.0f);
    #pragma unroll 8
    for (int i = 0; i < F_DIM; ++i) {
        s += __shfl(xv, i, 32) * root[i * F_DIM + lane];
    }
    xn[(size_t)n * F_DIM + lane] = fmaxf(s + bias[lane], 0.f);
}

// ---------------------------------------------------------------------------
// Kernel: final projection 32 -> 1
// ---------------------------------------------------------------------------
__global__ void out_kernel(const float* __restrict__ x, const float* __restrict__ W_out,
                           const float* __restrict__ b_out, float* __restrict__ out, int N) {
    int n = blockIdx.x * blockDim.x + threadIdx.x;
    if (n >= N) return;
    float s = b_out[0];
    #pragma unroll
    for (int o = 0; o < F_DIM; ++o) s += x[(size_t)n * F_DIM + o] * W_out[o];
    out[n] = s;
}

// ---------------------------------------------------------------------------
// Host launcher
// ---------------------------------------------------------------------------
extern "C" void kernel_launch(void* const* d_in, const int* in_sizes, int n_in,
                              void* d_out, int out_size, void* d_ws, size_t ws_size,
                              hipStream_t stream) {
    const float* xpos   = (const float*)d_in[0];
    const float* eattr  = (const float*)d_in[1];
    const float* W_ih   = (const float*)d_in[2];
    const float* W_hh   = (const float*)d_in[3];
    const float* b_ih   = (const float*)d_in[4];
    const float* b_hh   = (const float*)d_in[5];
    const float* W_fc   = (const float*)d_in[6];
    const float* b_fc   = (const float*)d_in[7];
    const float* kW1    = (const float*)d_in[8];
    const float* kb1    = (const float*)d_in[9];
    const float* kW2    = (const float*)d_in[10];
    const float* kb2    = (const float*)d_in[11];
    const float* kW3    = (const float*)d_in[12];
    const float* kb3    = (const float*)d_in[13];
    const float* root1  = (const float*)d_in[14];
    const float* bias1  = (const float*)d_in[15];
    const float* root2  = (const float*)d_in[16];
    const float* bias2  = (const float*)d_in[17];
    const float* W_out  = (const float*)d_in[18];
    const float* b_out  = (const float*)d_in[19];
    const int*   eidx   = (const int*)d_in[20];

    const int N = in_sizes[0] / (W_STEPS * 3);
    const int E = in_sizes[1] / KIN_DIM;

    // ---- workspace carve-up (256B aligned) -------------------------------
    char*  ws  = (char*)d_ws;
    size_t off = 0;
    auto carve = [&](size_t bytes) -> void* {
        void* p = ws + off;
        off += (bytes + 255) & ~(size_t)255;
        return p;
    };
    float*          x_a   = (float*)carve((size_t)N * F_DIM * sizeof(float));
    float*          x_b   = (float*)carve((size_t)N * F_DIM * sizeof(float));
    float*          agg   = (float*)carve((size_t)N * F_DIM * sizeof(float));
    float*          cnt   = (float*)carve((size_t)N * sizeof(float));
    unsigned short* kW2b  = (unsigned short*)carve((size_t)KW_DIM * KW_DIM * 2);
    unsigned short* kW3b  = (unsigned short*)carve((size_t)F_DIM * F_DIM * KW_DIM * 2);
    unsigned short* wbf   = (unsigned short*)carve((size_t)E * 1024 * 2);
    (void)ws_size;

    // ---- stage bf16 weights ----------------------------------------------
    {
        int n2 = KW_DIM * KW_DIM;
        int n3 = F_DIM * F_DIM * KW_DIM;
        cvt_bf16_kernel<<<(n2 + 255) / 256, 256, 0, stream>>>(kW2, kW2b, n2);
        cvt_bf16_kernel<<<(n3 + 255) / 256, 256, 0, stream>>>(kW3, kW3b, n3);
    }

    // ---- node features: LSTM + fc ----------------------------------------
    lstm_fc_kernel<<<(N + 127) / 128, 128, 0, stream>>>(
        xpos, W_ih, W_hh, b_ih, b_hh, W_fc, b_fc, x_a, N);

    // ---- degrees for scatter-mean ----------------------------------------
    hipMemsetAsync(cnt, 0, (size_t)N * sizeof(float), stream);
    degree_kernel<<<(E + 255) / 256, 256, 0, stream>>>(eidx, cnt, E);

    // ---- edge-conditioned kernels via WMMA bf16 --------------------------
    edge_mlp_kernel<<<(E + TILE_E - 1) / TILE_E, 256, 0, stream>>>(
        eattr, kW1, kb1, kb2, kb3, kW2b, kW3b, wbf, E);

    // ---- 2x conv1 + 2x conv2 ---------------------------------------------
    float* xc = x_a;
    float* xn = x_b;
    for (int it = 0; it < 4; ++it) {
        const float* root = (it < 2) ? root1 : root2;
        const float* bias = (it < 2) ? bias1 : bias2;
        hipMemsetAsync(agg, 0, (size_t)N * F_DIM * sizeof(float), stream);
        conv_msg_kernel<<<((size_t)E * 32 + 255) / 256, 256, 0, stream>>>(
            xc, wbf, eidx, agg, E);
        combine_kernel<<<((size_t)N * 32 + 255) / 256, 256, 0, stream>>>(
            xc, agg, cnt, root, bias, xn, N);
        float* t = xc; xc = xn; xn = t;
    }

    // ---- final projection -------------------------------------------------
    out_kernel<<<(N + 255) / 256, 256, 0, stream>>>(xc, W_out, b_out, (float*)d_out, N);
}
